// SPN_11467562680454
// MI455X (gfx1250) — matrix operand order repaired and checked by reading, not compiled, not run on hardware
//
#include <hip/hip_runtime.h>
#include <math.h>

// ---------------------------------------------------------------------------
// MI455X (gfx1250) implementation of the SPN inpainting network.
// All GEMM-like convolutions run on v_wmma_f32_16x16x32_bf16 (wave32 WMMA),
// 4 output tiles per wave for A-fragment reuse. SPN scans use per-thread
// float4 chunked loads (horizontal) / coalesced loads (vertical) + prefetch.
// ---------------------------------------------------------------------------

typedef __attribute__((ext_vector_type(16))) __bf16 v16bf;
typedef __attribute__((ext_vector_type(8)))  __bf16 v8bf;
typedef __attribute__((ext_vector_type(8)))  float  v8f;

static __device__ __forceinline__ __bf16 f2bf(float f) {
  unsigned u = __builtin_bit_cast(unsigned, f);
  unsigned r = u + 0x7fffu + ((u >> 16) & 1u);   // round-to-nearest-even
  return __builtin_bit_cast(__bf16, (unsigned short)(r >> 16));
}
static __device__ __forceinline__ __bf16 bf_zero() {
  return __builtin_bit_cast(__bf16, (unsigned short)0);
}

// ------------------------------ elementwise --------------------------------

__global__ void k_fill_f32(float* p, float v, size_t n) {
  size_t i = (size_t)blockIdx.x * blockDim.x + threadIdx.x;
  if (i < n) p[i] = v;
}

__global__ void k_f32_to_bf16(const float* __restrict__ in, __bf16* __restrict__ out, size_t n) {
  size_t i = (size_t)blockIdx.x * blockDim.x + threadIdx.x;
  if (i < n) out[i] = f2bf(in[i]);
}

// weights [Cout, K] f32 -> bf16 [CoutPad, K], zero-padded rows
__global__ void k_repack_w(const float* __restrict__ w, __bf16* __restrict__ out,
                           int Cout, int K, int CoutPad) {
  size_t n = (size_t)CoutPad * K;
  size_t i = (size_t)blockIdx.x * blockDim.x + threadIdx.x;
  if (i >= n) return;
  int row = (int)(i / K);
  int col = (int)(i - (size_t)row * K);
  out[i] = (row < Cout) ? f2bf(w[(size_t)row * K + col]) : bf_zero();
}

__global__ void k_pad_bias(const float* __restrict__ b, float* __restrict__ out,
                           int Cout, int CoutPad) {
  int i = blockIdx.x * blockDim.x + threadIdx.x;
  if (i < CoutPad) out[i] = (i < Cout) ? b[i] : 0.f;
}

// ------------------------ direct conv (Cin == 3) ---------------------------

__global__ void k_conv_direct(const float* __restrict__ X, const float* __restrict__ Wt,
                              const float* __restrict__ Bias, float* __restrict__ Y,
                              int Bn, int Cin, int H, int W, int Cout) {
  size_t total = (size_t)Bn * Cout * H * W;
  size_t i = (size_t)blockIdx.x * blockDim.x + threadIdx.x;
  if (i >= total) return;
  int w = (int)(i % W); size_t t = i / W;
  int h = (int)(t % H); t /= H;
  int co = (int)(t % Cout); int b = (int)(t / Cout);
  float acc = Bias[co];
  for (int ci = 0; ci < Cin; ++ci)
    for (int dy = -1; dy <= 1; ++dy)
      for (int dx = -1; dx <= 1; ++dx) {
        int y = h + dy, x = w + dx;
        if (y >= 0 && y < H && x >= 0 && x < W)
          acc += X[(((size_t)b * Cin + ci) * H + y) * W + x] *
                 Wt[(((size_t)co * Cin + ci) * 3 + (dy + 1)) * 3 + (dx + 1)];
      }
  Y[i] = acc;
}

// -------------------- maxpool(3,2,1) fused with ReLU -----------------------

__global__ void k_maxpool_relu_bf16(const float* __restrict__ X, __bf16* __restrict__ Y,
                                    int Bn, int C, int H, int W) {
  int Ho = H / 2, Wo = W / 2;
  size_t total = (size_t)Bn * C * Ho * Wo;
  size_t i = (size_t)blockIdx.x * blockDim.x + threadIdx.x;
  if (i >= total) return;
  int ox = (int)(i % Wo); size_t t = i / Wo;
  int oy = (int)(t % Ho); t /= Ho;
  int c = (int)(t % C); int b = (int)(t / C);
  const float* P = X + ((size_t)b * C + c) * H * W;
  float m = -3.4e38f;
  for (int ky = 0; ky < 3; ++ky)
    for (int kx = 0; kx < 3; ++kx) {
      int y = 2 * oy - 1 + ky, x = 2 * ox - 1 + kx;
      if (y >= 0 && y < H && x >= 0 && x < W) m = fmaxf(m, P[(size_t)y * W + x]);
    }
  Y[i] = f2bf(fmaxf(m, 0.f));  // max(relu(window)) == relu(max(window))
}

// ---------------- bilinear x2 upsample (half-pixel) -> bf16 ----------------

__global__ void k_up2_bf16(const float* __restrict__ X, __bf16* __restrict__ Y,
                           int Bn, int C, int H, int W) {
  int Ho = 2 * H, Wo = 2 * W;
  size_t total = (size_t)Bn * C * Ho * Wo;
  size_t i = (size_t)blockIdx.x * blockDim.x + threadIdx.x;
  if (i >= total) return;
  int ox = (int)(i % Wo); size_t t = i / Wo;
  int oy = (int)(t % Ho); t /= Ho;
  int c = (int)(t % C); int b = (int)(t / C);
  float fy = (oy + 0.5f) * 0.5f - 0.5f;
  float fx = (ox + 0.5f) * 0.5f - 0.5f;
  int y0 = (int)floorf(fy), x0 = (int)floorf(fx);
  float wy = fy - y0, wx = fx - x0;
  int y0c = min(max(y0, 0), H - 1), y1c = min(max(y0 + 1, 0), H - 1);
  int x0c = min(max(x0, 0), W - 1), x1c = min(max(x0 + 1, 0), W - 1);
  const float* P = X + ((size_t)b * C + c) * H * W;
  float v = (1.f - wy) * ((1.f - wx) * P[(size_t)y0c * W + x0c] + wx * P[(size_t)y0c * W + x1c]) +
            wy        * ((1.f - wx) * P[(size_t)y1c * W + x0c] + wx * P[(size_t)y1c * W + x1c]);
  Y[i] = f2bf(v);
}

// ---------------------- SPN gate normalization -----------------------------
// G: [Bn, 96, H, W]; triples (c, c+32, c+64)

__global__ void k_gatenorm(float* __restrict__ G, int Bn, int H, int W) {
  size_t total = (size_t)Bn * 32 * H * W;
  size_t i = (size_t)blockIdx.x * blockDim.x + threadIdx.x;
  if (i >= total) return;
  int w = (int)(i % W); size_t t = i / W;
  int h = (int)(t % H); t /= H;
  int c = (int)(t % 32); int b = (int)(t / 32);
  size_t plane = (size_t)32 * H * W;
  size_t p0 = (((size_t)b * 96 + c) * H + h) * W + w;
  float g1 = G[p0], g2 = G[p0 + plane], g3 = G[p0 + 2 * plane];
  float s = fabsf(g1) + fabsf(g2) + fabsf(g3);
  if (s == 0.f) s = 1e-6f;
  if (s >= 1.f) {
    G[p0] = g1 / s; G[p0 + plane] = g2 / s; G[p0 + 2 * plane] = g3 / s;
  }
}

// -------------------------- SPN recurrent scan -----------------------------
// One block per (b,c); 256 threads = rows; 256 sequential steps via LDS.
// Horizontal: each thread's row is contiguous -> float4 chunked loads.
// Vertical: per-step loads are coalesced across threads -> scalar + prefetch.

__global__ void k_spn_scan(const float* __restrict__ X, const float* __restrict__ G,
                           float* __restrict__ Omax,
                           int H, int W, int horizontal, int reverse) {
  __shared__ float hbuf[258];
  int bc = blockIdx.x;
  int b = bc >> 5, c = bc & 31;
  int L = horizontal ? H : W;   // parallel (row) dimension
  int T = horizontal ? W : H;   // scan dimension
  int i = threadIdx.x;
  if (i == 0) { hbuf[0] = 0.f; hbuf[L + 1] = 0.f; }
  hbuf[i + 1] = 0.f;
  __syncthreads();
  size_t xbase = (((size_t)b * 32 + c) * H) * (size_t)W;
  size_t gbase = (((size_t)b * 96 + c) * H) * (size_t)W;
  size_t gplane = (size_t)32 * H * W;

  auto step = [&](float xv, float g1, float g2, float g3, size_t ooff) {
    float hu = hbuf[i];
    float hc = hbuf[i + 1];
    float hd = hbuf[i + 2];
    float hn = (1.f - g1 - g2 - g3) * xv + g1 * hu + g2 * hc + g3 * hd;
    __syncthreads();
    hbuf[i + 1] = hn;
    float* op = Omax + ooff;
    *op = fmaxf(*op, hn);
    __syncthreads();
  };

  if (horizontal) {
    const float* xr  = X + xbase + (size_t)i * W;
    const float* g1r = G + gbase + (size_t)i * W;
    const float* g2r = g1r + gplane;
    const float* g3r = g2r + gplane;
    size_t obase = xbase + (size_t)i * W;
    for (int t0 = 0; t0 < T; t0 += 4) {
      int wb = reverse ? (W - 4 - t0) : t0;           // 16B-aligned chunk base
      if (t0 + 4 < T) {
        int wn = reverse ? (wb - 4) : (wb + 4);
        __builtin_prefetch(xr + wn, 0, 0);
        __builtin_prefetch(g1r + wn, 0, 0);
        __builtin_prefetch(g2r + wn, 0, 0);
        __builtin_prefetch(g3r + wn, 0, 0);
      }
      float4 x4 = *(const float4*)(xr + wb);
      float4 a4 = *(const float4*)(g1r + wb);
      float4 b4 = *(const float4*)(g2r + wb);
      float4 c4 = *(const float4*)(g3r + wb);
      if (!reverse) {
        step(x4.x, a4.x, b4.x, c4.x, obase + wb + 0);
        step(x4.y, a4.y, b4.y, c4.y, obase + wb + 1);
        step(x4.z, a4.z, b4.z, c4.z, obase + wb + 2);
        step(x4.w, a4.w, b4.w, c4.w, obase + wb + 3);
      } else {
        step(x4.w, a4.w, b4.w, c4.w, obase + wb + 3);
        step(x4.z, a4.z, b4.z, c4.z, obase + wb + 2);
        step(x4.y, a4.y, b4.y, c4.y, obase + wb + 1);
        step(x4.x, a4.x, b4.x, c4.x, obase + wb + 0);
      }
    }
  } else {
    for (int t = 0; t < T; ++t) {
      int hh = reverse ? (H - 1 - t) : t;
      size_t off = (size_t)hh * W + i;
      if (t + 1 < T) {
        int hn2 = reverse ? (hh - 1) : (hh + 1);
        __builtin_prefetch(X + xbase + (size_t)hn2 * W + i, 0, 0);
        __builtin_prefetch(G + gbase + (size_t)hn2 * W + i, 0, 0);
      }
      float xv = X[xbase + off];
      float g1 = G[gbase + off];
      float g2 = G[gbase + gplane + off];
      float g3 = G[gbase + 2 * gplane + off];
      step(xv, g1, g2, g3, xbase + off);
    }
  }
}

// ------------------------- WMMA implicit-GEMM conv -------------------------
// One wave (32 threads) per 16(cout) x 64(pixel) macro-tile (4 N-tiles).
// A = weights bf16 [CoutPad, K] (K = Cin*KSZ, multiple of 32) -> b128 loads.
// B = on-the-fly im2col gather (clamped address + cndmask, no exec branches).

template <int KSZ>
__global__ void k_conv_wmma(const __bf16* __restrict__ X, const __bf16* __restrict__ Wt,
                            const float* __restrict__ BiasP,   // padded to CoutPad
                            float* __restrict__ Yf, __bf16* __restrict__ Yh,
                            const float* __restrict__ Res,
                            int Cin, int H, int W, int Cout, int doElu) {
  const int K = Cin * KSZ;
  const int lane = threadIdx.x;
  const int half = lane >> 4;
  const int l15 = lane & 15;
  const int tileN0 = blockIdx.x << 6;          // 64 pixels per wave
  const int tileM = blockIdx.y << 4;
  const int HW = H * W;

  int bs[4], pys[4], pxws[4];
#pragma unroll
  for (int t = 0; t < 4; ++t) {
    int px = tileN0 + (t << 4) + l15;          // flat pixel id in [Bn*H*W)
    int b = px / HW;
    int rr = px - b * HW;
    int py = rr / W;
    bs[t] = b; pys[t] = py; pxws[t] = rr - py * W;
  }

  // unconditional bias init (BiasP is zero-padded to CoutPad)
  v8f acc[4];
  {
    const int rbase = tileM + half * 8;
#pragma unroll
    for (int v = 0; v < 8; ++v) {
      float bv = BiasP[rbase + v];
#pragma unroll
      for (int t = 0; t < 4; ++t) acc[t][v] = bv;
    }
  }

  const __bf16* wrow = Wt + (size_t)(tileM + l15) * K;
  for (int kb = 0; kb < K; kb += 32) {
    // ----- A fragment: 16-bit A layout (lanes 0-15: K 0..7 / 16..23;
    //                    lanes 16-31: K 8..15 / 24..31); two b128 loads.
    const __bf16* wp = wrow + kb + half * 8;
    __builtin_prefetch(wp + 128, 0, 1);
    v8bf lo = *(const v8bf*)(wp);
    v8bf hi = *(const v8bf*)(wp + 16);
    v16bf a;
#pragma unroll
    for (int j = 0; j < 8; ++j) { a[j] = lo[j]; a[8 + j] = hi[j]; }

    // ----- B fragments for the 4 N-tiles; one WMMA each (A reused 4x).
    const int kstart = kb + half * 8;
#pragma unroll
    for (int t = 0; t < 4; ++t) {
      const int b = bs[t], py = pys[t], pxw = pxws[t];
      const size_t xb = (size_t)b * Cin * HW;
      v16bf bm;
#pragma unroll
      for (int j = 0; j < 16; ++j) {
        int k = kstart + j + ((j >> 3) << 3);
        int ci, dy, dx;
        if (KSZ == 9) {
          ci = k / 9;
          int rs = k - ci * 9;
          dy = rs / 3 - 1;
          dx = rs - (rs / 3) * 3 - 1;
        } else {
          ci = k; dy = 0; dx = 0;
        }
        int sy = py + dy, sx = pxw + dx;
        // clamped unconditional load + select (avoids exec-mask branches)
        int syc = min(max(sy, 0), H - 1);
        int sxc = min(max(sx, 0), W - 1);
        __bf16 v = X[xb + ((size_t)ci * H + syc) * W + sxc];
        bool inb = (sy == syc) & (sx == sxc);
        bm[j] = inb ? v : bf_zero();
      }
      acc[t] = __builtin_amdgcn_wmma_f32_16x16x32_bf16(
          false, a, false, bm, (short)0, acc[t], false, false);
    }
  }

#pragma unroll
  for (int t = 0; t < 4; ++t) {
    const int b = bs[t], py = pys[t], pxw = pxws[t];
#pragma unroll
    for (int v = 0; v < 8; ++v) {
      int row = tileM + half * 8 + v;
      if (row < Cout) {
        float val = acc[t][v];
        size_t idx = (((size_t)b * Cout + row) * H + py) * W + pxw;
        if (doElu) val = (val > 0.f) ? val : (__expf(val) - 1.f);
        if (Res) val += Res[idx];
        if (Yf) Yf[idx] = val;
        if (Yh) Yh[idx] = f2bf(val);
      }
    }
  }
}

// ---------------------------------------------------------------------------

extern "C" void kernel_launch(void* const* d_in, const int* in_sizes, int n_in,
                              void* d_out, int out_size, void* d_ws, size_t ws_size,
                              hipStream_t stream) {
  (void)in_sizes; (void)n_in; (void)out_size; (void)ws_size;
  const int B = 2, C = 32, H0 = 256, W0 = 256;

  int idx = 0;
  const float* mask   = (const float*)d_in[idx++];
  const float* rgb    = (const float*)d_in[idx++];
  const float* mask_w = (const float*)d_in[idx++];
  const float* mask_b = (const float*)d_in[idx++];
  const float* e1w = (const float*)d_in[idx++]; const float* e1b = (const float*)d_in[idx++];
  const float* e2w = (const float*)d_in[idx++]; const float* e2b = (const float*)d_in[idx++];
  const float* e3w = (const float*)d_in[idx++]; const float* e3b = (const float*)d_in[idx++];
  const float* e4w = (const float*)d_in[idx++]; const float* e4b = (const float*)d_in[idx++];
  const float* d0w = (const float*)d_in[idx++]; const float* d0b = (const float*)d_in[idx++];
  const float* d2w = (const float*)d_in[idx++]; const float* d2b = (const float*)d_in[idx++];
  const float* d4w = (const float*)d_in[idx++]; const float* d4b = (const float*)d_in[idx++];
  const float* d6w = (const float*)d_in[idx++]; const float* d6b = (const float*)d_in[idx++];
  const float* d7w = (const float*)d_in[idx++]; const float* d7b = (const float*)d_in[idx++];
  const float* postw = (const float*)d_in[idx++]; const float* postb = (const float*)d_in[idx++];

  // ------------------------- workspace carving -----------------------------
  char* ws = (char*)d_ws;
  size_t cur = 0;
  auto carve = [&](size_t bytes) -> void* {
    void* p = ws + cur;
    cur += (bytes + 255) & ~(size_t)255;
    return p;
  };
  auto f32buf  = [&](size_t elems) { return (float*) carve(elems * 4); };
  auto bf16buf = [&](size_t elems) { return (__bf16*)carve(elems * 2); };

  // bf16 weights (rows padded to 16)
  __bf16* wb_e2   = bf16buf((size_t)64  * 288);
  __bf16* wb_e3   = bf16buf((size_t)128 * 576);
  __bf16* wb_e4   = bf16buf((size_t)256 * 1152);
  __bf16* wb_d0   = bf16buf((size_t)128 * 256);
  __bf16* wb_d2   = bf16buf((size_t)128 * 1152);
  __bf16* wb_d4   = bf16buf((size_t)64  * 1152);
  __bf16* wb_d6   = bf16buf((size_t)32  * 576);
  __bf16* wb_d7   = bf16buf((size_t)384 * 288);
  __bf16* wb_post = bf16buf((size_t)16  * 288);

  // zero-padded f32 biases
  float* bp_e2   = f32buf(64);
  float* bp_e3   = f32buf(128);
  float* bp_e4   = f32buf(256);
  float* bp_d0   = f32buf(128);
  float* bp_d2   = f32buf(128);
  float* bp_d4   = f32buf(64);
  float* bp_d6   = f32buf(32);
  float* bp_d7   = f32buf(384);
  float* bp_post = f32buf(16);

  // activations
  float*  x_spn = f32buf((size_t)B * 32 * 256 * 256);
  float*  c1    = f32buf((size_t)B * 32 * 256 * 256);
  __bf16* p1    = bf16buf((size_t)B * 32 * 128 * 128);
  float*  c2    = f32buf((size_t)B * 64 * 128 * 128);
  __bf16* p2    = bf16buf((size_t)B * 64 * 64 * 64);
  float*  c3    = f32buf((size_t)B * 128 * 64 * 64);
  __bf16* p3    = bf16buf((size_t)B * 128 * 32 * 32);
  __bf16* c4h   = bf16buf((size_t)B * 256 * 32 * 32);
  float*  t0    = f32buf((size_t)B * 128 * 32 * 32);
  __bf16* u1    = bf16buf((size_t)B * 128 * 64 * 64);
  float*  o3    = f32buf((size_t)B * 128 * 64 * 64);
  __bf16* u2    = bf16buf((size_t)B * 128 * 128 * 128);
  float*  o2    = f32buf((size_t)B * 64 * 128 * 128);
  __bf16* u3    = bf16buf((size_t)B * 64 * 256 * 256);
  float*  o1    = f32buf((size_t)B * 32 * 256 * 256);
  __bf16* o1h   = bf16buf((size_t)B * 32 * 256 * 256);
  float*  guide = f32buf((size_t)B * 96 * 256 * 256);   // one direction at a time
  float*  sm    = f32buf((size_t)B * 32 * 256 * 256);
  __bf16* smh   = bf16buf((size_t)B * 32 * 256 * 256);

  // ------------------------------ helpers ----------------------------------
  auto grid1 = [](size_t n) { return dim3((unsigned)((n + 255) / 256)); };

  auto repack = [&](const float* w, const float* b, __bf16* wdst, float* bdst,
                    int Cout, int K) {
    int CoutPad = (Cout + 15) & ~15;
    size_t n = (size_t)CoutPad * K;
    k_repack_w<<<grid1(n), dim3(256), 0, stream>>>(w, wdst, Cout, K, CoutPad);
    k_pad_bias<<<dim3(1), dim3(((CoutPad + 63) / 64) * 64), 0, stream>>>(b, bdst, Cout, CoutPad);
  };

  auto convw = [&](const __bf16* X, const __bf16* Wt, const float* biasP,
                   float* Yf, __bf16* Yh, const float* Res,
                   int Cin, int H, int W, int Cout, int ksz, int doElu) {
    int NP = B * H * W;
    dim3 grid((unsigned)(NP / 64), (unsigned)((Cout + 15) / 16));
    if (ksz == 9)
      k_conv_wmma<9><<<grid, dim3(32), 0, stream>>>(X, Wt, biasP, Yf, Yh, Res,
                                                    Cin, H, W, Cout, doElu);
    else
      k_conv_wmma<1><<<grid, dim3(32), 0, stream>>>(X, Wt, biasP, Yf, Yh, Res,
                                                    Cin, H, W, Cout, doElu);
  };

  // --------------------------- weight repacks ------------------------------
  repack(e2w, e2b, wb_e2, bp_e2, 64, 288);
  repack(e3w, e3b, wb_e3, bp_e3, 128, 576);
  repack(e4w, e4b, wb_e4, bp_e4, 256, 1152);
  repack(d0w, d0b, wb_d0, bp_d0, 128, 256);
  repack(d2w, d2b, wb_d2, bp_d2, 128, 1152);
  repack(d4w, d4b, wb_d4, bp_d4, 64, 1152);
  repack(d6w, d6b, wb_d6, bp_d6, 32, 576);
  repack(d7w, d7b, wb_d7, bp_d7, 384, 288);
  repack(postw, postb, wb_post, bp_post, 3, 288);

  // ------------------------------ network ----------------------------------
  // mask head + encoder entry (Cin=3 -> direct conv)
  {
    size_t n = (size_t)B * C * H0 * W0;
    k_conv_direct<<<grid1(n), dim3(256), 0, stream>>>(mask, mask_w, mask_b, x_spn, B, 3, H0, W0, C);
    k_conv_direct<<<grid1(n), dim3(256), 0, stream>>>(rgb, e1w, e1b, c1, B, 3, H0, W0, C);
  }
  // encoder
  k_maxpool_relu_bf16<<<grid1((size_t)B * 32 * 128 * 128), dim3(256), 0, stream>>>(c1, p1, B, 32, 256, 256);
  convw(p1, wb_e2, bp_e2, c2, nullptr, nullptr, 32, 128, 128, 64, 9, 0);
  k_maxpool_relu_bf16<<<grid1((size_t)B * 64 * 64 * 64), dim3(256), 0, stream>>>(c2, p2, B, 64, 128, 128);
  convw(p2, wb_e3, bp_e3, c3, nullptr, nullptr, 64, 64, 64, 128, 9, 0);
  k_maxpool_relu_bf16<<<grid1((size_t)B * 128 * 32 * 32), dim3(256), 0, stream>>>(c3, p3, B, 128, 64, 64);
  convw(p3, wb_e4, bp_e4, nullptr, c4h, nullptr, 128, 32, 32, 256, 9, 0);
  // decoder
  convw(c4h, wb_d0, bp_d0, t0, nullptr, nullptr, 256, 32, 32, 128, 1, 0);
  k_up2_bf16<<<grid1((size_t)B * 128 * 64 * 64), dim3(256), 0, stream>>>(t0, u1, B, 128, 32, 32);
  convw(u1, wb_d2, bp_d2, o3, nullptr, c3, 128, 64, 64, 128, 9, 1);
  k_up2_bf16<<<grid1((size_t)B * 128 * 128 * 128), dim3(256), 0, stream>>>(o3, u2, B, 128, 64, 64);
  convw(u2, wb_d4, bp_d4, o2, nullptr, c2, 128, 128, 128, 64, 9, 1);
  k_up2_bf16<<<grid1((size_t)B * 64 * 256 * 256), dim3(256), 0, stream>>>(o2, u3, B, 64, 128, 128);
  convw(u3, wb_d6, bp_d6, o1, o1h, c1, 64, 256, 256, 32, 9, 1);

  // SPN: per direction, compute a 96-channel guide slice, normalize, scan,
  // max-accumulate into sm.
  k_fill_f32<<<grid1((size_t)B * 32 * 256 * 256), dim3(256), 0, stream>>>(sm, -3.4e38f, (size_t)B * 32 * 256 * 256);
  for (int d = 0; d < 4; ++d) {
    convw(o1h, wb_d7 + (size_t)d * 96 * 288, bp_d7 + d * 96, guide, nullptr, nullptr,
          32, 256, 256, 96, 9, 0);
    k_gatenorm<<<grid1((size_t)B * 32 * 256 * 256), dim3(256), 0, stream>>>(guide, B, 256, 256);
    int horizontal = (d < 2) ? 1 : 0;
    int reverse = d & 1;
    k_spn_scan<<<dim3(B * 32), dim3(256), 0, stream>>>(x_spn, guide, sm, 256, 256, horizontal, reverse);
  }

  // post conv (Cout=3, rows zero-padded to 16 in wb_post / bp_post)
  k_f32_to_bf16<<<grid1((size_t)B * 32 * 256 * 256), dim3(256), 0, stream>>>(sm, smh, (size_t)B * 32 * 256 * 256);
  convw(smh, wb_post, bp_post, (float*)d_out, nullptr, nullptr, 32, 256, 256, 3, 9, 0);
}
